// PoseTransformer_66829691126076
// MI455X (gfx1250) — matrix operand order repaired and checked by reading, not compiled, not run on hardware
//
#include <hip/hip_runtime.h>

#define DEV __device__ __forceinline__

typedef __attribute__((ext_vector_type(16))) __bf16 v16bf;
typedef __attribute__((ext_vector_type(8)))  float  v8f;
typedef __attribute__((ext_vector_type(4)))  float  f32x4;

namespace {

constexpr int E  = 64;
constexpr int H  = 4;
constexpr int FFD = 32;
constexpr int L  = 9;
constexpr int S  = 7;
constexpr int HD = 16;

constexpr int WAVES_PER_BLOCK = 4;
constexpr int ELEMS_PER_WAVE  = 2;
constexpr int ELEMS_PER_BLOCK = WAVES_PER_BLOCK * ELEMS_PER_WAVE; // 8
constexpr int THREADS = WAVES_PER_BLOCK * 32;

// Per-wave LDS working set (floats). Logical rows are (elem, s) pairs: 14 rows.
constexpr int NF_ACT = ELEMS_PER_WAVE * S * E;       // 896
constexpr int NF_KV  = ELEMS_PER_WAVE * 2 * S * E;   // 1792 (cross-attn K/V)
constexpr int NF_H   = ELEMS_PER_WAVE * S * FFD;     // 448

struct WaveBuf {
  float f1 [NF_ACT];
  float f2 [NF_ACT];
  float q  [NF_ACT];
  float pq [NF_ACT];
  float pk [NF_KV];
  float pv [NF_KV];
  float ctx[NF_ACT];
  float h  [NF_H];
  float zrow[64];   // zero row: A-operand source for padding rows (no branches)
  float dump[64];   // scratch row: D writeback target for padding rows
};

// ---------------------------------------------------------------------------
// WMMA GEMM: LDS f32 activations (A, converted to bf16 in-regs) x global bf16
// weights (B).  y[m][n] = sum_k x[m][k] * W[n][k]  (+bias, opt. relu/residual)
//
// A fragment (16x32 bf16, ISA 7.12.2): lane l, m = l&15, kh = l>>4;
//   elems i <- K = k0 + kh*16 + i  (contiguous order, chosen here to match B)
// B fragment (32x16 bf16): lane l, n = n0 + (l&15), kh = l>>4;
//   elems i <- K = k0 + kh*16 + i  (16 contiguous bf16 of W row n -> b128 loads)
// C/D (16x16 f32): lane l, col n = n0 + (l&15); VGPR v -> row kh*8 + v.
//
// arow(r)/drow(r,n0) ALWAYS return valid LDS pointers (pad rows -> zrow/dump):
// every load/store is unconditional (no EXEC save/restore).
// B fragments are software-pipelined: tile j+1's global loads issue before
// tile j's WMMAs so VMEM latency overlaps the matrix pipe. Biases are hoisted
// into registers before the loop.
// ---------------------------------------------------------------------------
template <int MT, int NT, int KT, bool RES, bool RELU, typename ARowF, typename DRowF>
DEV void wave_gemm(int lane, ARowF arow, DRowF drow,
                   const __bf16* __restrict__ W,
                   const float*  __restrict__ bias)
{
  const int nl = lane & 15;
  const int kh = lane >> 4;
  constexpr int KFRAG  = KT / 32;
  constexpr int NTILES = NT / 16;
  for (int mt = 0; mt < MT; ++mt) {
    const float* arp = arow(mt * 16 + nl);     // always valid
    // A fragments: load once per (mt, k0); reused across all N tiles.
    v16bf afrag[KFRAG];
    #pragma unroll
    for (int kf = 0; kf < KFRAG; ++kf) {
      const float* a0 = arp + kf * 32 + kh * 16;
      f32x4 x0 = *(const f32x4*)(a0 + 0);
      f32x4 x1 = *(const f32x4*)(a0 + 4);
      f32x4 x2 = *(const f32x4*)(a0 + 8);
      f32x4 x3 = *(const f32x4*)(a0 + 12);
      v16bf a;
      #pragma unroll
      for (int i = 0; i < 4; ++i) {
        a[i]      = (__bf16)x0[i];
        a[i + 4]  = (__bf16)x1[i];
        a[i + 8]  = (__bf16)x2[i];
        a[i + 12] = (__bf16)x3[i];
      }
      afrag[kf] = a;
    }
    // Hoist bias scalars out of the hot loop.
    float bv[NTILES];
    #pragma unroll
    for (int j = 0; j < NTILES; ++j) bv[j] = bias[j * 16 + nl];

    // Prologue: prefetch B fragments for tile 0.
    v16bf bnext[KFRAG];
    {
      const __bf16* wrow = W + (size_t)nl * KT;
      #pragma unroll
      for (int kf = 0; kf < KFRAG; ++kf)
        bnext[kf] = *(const v16bf*)(wrow + kf * 32 + kh * 16);
    }
    #pragma unroll
    for (int j = 0; j < NTILES; ++j) {
      v16bf bcur[KFRAG];
      #pragma unroll
      for (int kf = 0; kf < KFRAG; ++kf) bcur[kf] = bnext[kf];
      if (j + 1 < NTILES) {                   // prefetch tile j+1 before using j
        const __bf16* wrow = W + (size_t)((j + 1) * 16 + nl) * KT;
        #pragma unroll
        for (int kf = 0; kf < KFRAG; ++kf)
          bnext[kf] = *(const v16bf*)(wrow + kf * 32 + kh * 16);
      }
      v8f acc = {};
      #pragma unroll
      for (int kf = 0; kf < KFRAG; ++kf)
        acc = __builtin_amdgcn_wmma_f32_16x16x32_bf16(false, afrag[kf], false, bcur[kf],
                                                      (short)0, acc, false, false);
      #pragma unroll
      for (int v = 0; v < 8; ++v) {
        const int r = mt * 16 + kh * 8 + v;
        float* dp = drow(r, j * 16);           // always valid; dp[nl] is target
        float val = acc[v] + bv[j];
        if (RELU) val = fmaxf(val, 0.0f);
        if (RES)  val += dp[nl];
        dp[nl] = val;
      }
    }
  }
}

// Scalar softmax-attention: 56 (elem, head, query) tasks across 32 lanes.
// All LDS accesses are f32x4 (64B-aligned offsets) -> ds_load/store_b128.
DEV void wave_attention(int lane, const float* pq, const float* pk, const float* pv,
                        float* ctx, int nk, int krows)
{
  for (int t = lane; t < ELEMS_PER_WAVE * H * S; t += 32) {
    const int e  = t / (H * S);
    const int rm = t - e * (H * S);
    const int hh = rm / S;
    const int sq = rm - hh * S;
    const f32x4* qp = (const f32x4*)(pq + (e * S + sq) * E + hh * HD);
    const f32x4 q0 = qp[0], q1 = qp[1], q2 = qp[2], q3 = qp[3];

    float sc[2 * S];
    float mx = -3.0e38f;
    for (int sk = 0; sk < nk; ++sk) {
      const f32x4* kp = (const f32x4*)(pk + (e * krows + sk) * E + hh * HD);
      f32x4 pr = q0 * kp[0] + q1 * kp[1] + q2 * kp[2] + q3 * kp[3];
      float d = (pr[0] + pr[1] + pr[2] + pr[3]) * 0.25f;   // 1/sqrt(HD=16)
      sc[sk] = d;
      mx = fmaxf(mx, d);
    }
    float sum = 0.0f;
    for (int sk = 0; sk < nk; ++sk) { float ex = __expf(sc[sk] - mx); sc[sk] = ex; sum += ex; }
    const float inv = 1.0f / sum;

    f32x4 a0 = {}, a1 = {}, a2 = {}, a3 = {};
    for (int sk = 0; sk < nk; ++sk) {
      const float w = sc[sk];
      const f32x4* vp = (const f32x4*)(pv + (e * krows + sk) * E + hh * HD);
      a0 += w * vp[0]; a1 += w * vp[1]; a2 += w * vp[2]; a3 += w * vp[3];
    }
    f32x4* cp = (f32x4*)(ctx + (e * S + sq) * E + hh * HD);
    cp[0] = a0 * inv; cp[1] = a1 * inv; cp[2] = a2 * inv; cp[3] = a3 * inv;
  }
}

DEV void self_attn(int lane, float* f, WaveBuf* wb,
                   const __bf16* win, const float* bin,
                   const __bf16* wout, const float* bout)
{
  // fused QKV projection: 14 rows x 192 cols, K=64
  auto arow = [&](int r) -> const float* { return (r < 2 * S) ? f + r * E : wb->zrow; };
  auto drow = [&](int r, int n0) -> float* {
    const int c = n0 >> 6, nl0 = n0 & 63;
    float* base = (c == 0) ? wb->pq : (c == 1 ? wb->pk : wb->pv);
    return (r < 2 * S) ? base + r * E + nl0 : wb->dump + nl0;
  };
  wave_gemm<1, 3 * E, E, false, false>(lane, arow, drow, win, bin);
  __syncthreads();

  wave_attention(lane, wb->pq, wb->pk, wb->pv, wb->ctx, S, S);
  __syncthreads();

  auto arow2 = [&](int r) -> const float* { return (r < 2 * S) ? wb->ctx + r * E : wb->zrow; };
  auto drow2 = [&](int r, int n0) -> float* { return (r < 2 * S) ? f + r * E + n0 : wb->dump + n0; };
  wave_gemm<1, E, E, true, false>(lane, arow2, drow2, wout, bout);   // residual add
  __syncthreads();
}

DEV void cross_attn(int lane, WaveBuf* wb,
                    const __bf16* win, const float* bin,
                    const __bf16* wout, const float* bout)
{
  // Q projection from q (rows of win[0:64])
  auto arowq = [&](int r) -> const float* { return (r < 2 * S) ? wb->q + r * E : wb->zrow; };
  auto drowq = [&](int r, int n0) -> float* { return (r < 2 * S) ? wb->pq + r * E + n0 : wb->dump + n0; };
  wave_gemm<1, E, E, false, false>(lane, arowq, drowq, win, bin);

  // K/V projections from concat(f1,f2): one 16-row M-tile per element (14 valid)
  auto arowkv = [&](int r) -> const float* {
    const int t = r >> 4, w = r & 15;
    if (w >= 2 * S) return wb->zrow;
    return (w < S) ? wb->f1 + (t * S + w) * E : wb->f2 + (t * S + (w - S)) * E;
  };
  auto drowk = [&](int r, int n0) -> float* {
    const int t = r >> 4, w = r & 15;
    return (w < 2 * S) ? wb->pk + (t * 2 * S + w) * E + n0 : wb->dump + n0;
  };
  auto drowv = [&](int r, int n0) -> float* {
    const int t = r >> 4, w = r & 15;
    return (w < 2 * S) ? wb->pv + (t * 2 * S + w) * E + n0 : wb->dump + n0;
  };
  wave_gemm<2, E, E, false, false>(lane, arowkv, drowk, win + (size_t)E * E,     bin + E);
  wave_gemm<2, E, E, false, false>(lane, arowkv, drowv, win + (size_t)2 * E * E, bin + 2 * E);
  __syncthreads();

  wave_attention(lane, wb->pq, wb->pk, wb->pv, wb->ctx, 2 * S, 2 * S);
  __syncthreads();

  auto arowc = [&](int r) -> const float* { return (r < 2 * S) ? wb->ctx + r * E : wb->zrow; };
  auto drowo = [&](int r, int n0) -> float* { return (r < 2 * S) ? wb->q + r * E + n0 : wb->dump + n0; };
  wave_gemm<1, E, E, true, false>(lane, arowc, drowo, wout, bout);   // residual add
  __syncthreads();
}

DEV void ffn(int lane, WaveBuf* wb, const __bf16* w1, const float* b1,
             const __bf16* w2, const float* b2)
{
  auto arow  = [&](int r) -> const float* { return (r < 2 * S) ? wb->q + r * E : wb->zrow; };
  auto drowh = [&](int r, int n0) -> float* { return (r < 2 * S) ? wb->h + r * FFD + n0 : wb->dump + n0; };
  wave_gemm<1, FFD, E, false, true>(lane, arow, drowh, w1, b1);      // relu
  __syncthreads();

  auto arowh = [&](int r) -> const float* { return (r < 2 * S) ? wb->h + r * FFD : wb->zrow; };
  auto drowq = [&](int r, int n0) -> float* { return (r < 2 * S) ? wb->q + r * E + n0 : wb->dump + n0; };
  wave_gemm<1, E, FFD, true, false>(lane, arowh, drowq, w2, b2);     // residual add
  __syncthreads();
}

struct Params {
  const float* feature1;
  const float* feature2;
  const float* query_embed;
  const __bf16 *w_sa1_in, *w_sa1_out, *w_sa2_in, *w_sa2_out;
  const __bf16 *w_ca_in,  *w_ca_out,  *w_ff1,    *w_ff2;
  const float  *b_sa1_in, *b_sa1_out, *b_sa2_in, *b_sa2_out;
  const float  *b_ca_in,  *b_ca_out,  *b_ff1,    *b_ff2;
  const float  *fc_rot_w, *fc_rot_b, *fc_trans_w, *fc_trans_b;
  float* out;
};

} // namespace

__global__ __launch_bounds__(THREADS)
void pose_transformer_kernel(Params p)
{
  extern __shared__ __align__(64) char smem[];
  const int wave = threadIdx.x >> 5;
  const int lane = threadIdx.x & 31;
  WaveBuf* wb = reinterpret_cast<WaveBuf*>(smem) + wave;
  const int bi = blockIdx.y;                                   // branch (rot / trans)
  const int eb = (blockIdx.x * WAVES_PER_BLOCK + wave) * ELEMS_PER_WAVE;

  // Stage activations into LDS (f32 master copies for residual fidelity).
  for (int idx = lane; idx < NF_ACT; idx += 32) {
    const int e = idx / (S * E);
    const int off = idx - e * (S * E);
    wb->f1[idx] = p.feature1[(size_t)(eb + e) * (S * E) + off];
    wb->f2[idx] = p.feature2[(size_t)(eb + e) * (S * E) + off];
    wb->q [idx] = p.query_embed[(size_t)bi * (S * E) + off];
  }
  for (int idx = lane; idx < 64; idx += 32) wb->zrow[idx] = 0.0f;
  __syncthreads();

  for (int i = 0; i < L; ++i) {
    const size_t li = (size_t)bi * L + i;
    if (i + 1 < L && lane == 0) {           // warm caches for next layer's weights
      __builtin_prefetch(p.w_sa1_in + (li + 1) * (size_t)(3 * E * E), 0, 1);
      __builtin_prefetch(p.w_sa2_in + (li + 1) * (size_t)(3 * E * E), 0, 1);
      __builtin_prefetch(p.w_ca_in  + (li + 1) * (size_t)(3 * E * E), 0, 1);
    }
    self_attn(lane, wb->f1, wb,
              p.w_sa1_in  + li * (size_t)(3 * E * E), p.b_sa1_in  + li * (3 * E),
              p.w_sa1_out + li * (size_t)(E * E),     p.b_sa1_out + li * E);
    self_attn(lane, wb->f2, wb,
              p.w_sa2_in  + li * (size_t)(3 * E * E), p.b_sa2_in  + li * (3 * E),
              p.w_sa2_out + li * (size_t)(E * E),     p.b_sa2_out + li * E);
    cross_attn(lane, wb,
              p.w_ca_in   + li * (size_t)(3 * E * E), p.b_ca_in   + li * (3 * E),
              p.w_ca_out  + li * (size_t)(E * E),     p.b_ca_out  + li * E);
    ffn(lane, wb,
        p.w_ff1 + li * (size_t)(FFD * E), p.b_ff1 + li * FFD,
        p.w_ff2 + li * (size_t)(E * FFD), p.b_ff2 + li * E);
  }

  // Final linear head + mean over S. Branch 0 -> cols 0..3, branch 1 -> cols 4..6.
  const float* fw = bi ? p.fc_trans_w : p.fc_rot_w;
  const float* fb = bi ? p.fc_trans_b : p.fc_rot_b;
  const int nout = bi ? 3 : 4;
  const int col0 = bi ? 4 : 0;
  for (int t = lane; t < ELEMS_PER_WAVE * nout; t += 32) {
    const int e = t / nout, j = t - e * nout;
    f32x4 acc = {};
    for (int s = 0; s < S; ++s) {
      const f32x4* qp = (const f32x4*)(wb->q + (e * S + s) * E);
      const f32x4* wp = (const f32x4*)(fw + j * E);
      #pragma unroll
      for (int k4 = 0; k4 < E / 4; ++k4) acc += qp[k4] * wp[k4];
    }
    const float tot = acc[0] + acc[1] + acc[2] + acc[3];
    p.out[(size_t)(eb + e) * 7 + col0 + j] = tot * (1.0f / 7.0f) + fb[j];
  }
}

__global__ void f32_to_bf16_kernel(const float* __restrict__ src,
                                   __bf16* __restrict__ dst, int n)
{
  const int i = blockIdx.x * blockDim.x + threadIdx.x;
  if (i < n) dst[i] = (__bf16)src[i];
}

extern "C" void kernel_launch(void* const* d_in, const int* in_sizes, int n_in,
                              void* d_out, int out_size, void* d_ws, size_t ws_size,
                              hipStream_t stream)
{
  (void)n_in; (void)out_size; (void)ws_size;

  const float* feature1    = (const float*)d_in[0];
  const float* feature2    = (const float*)d_in[1];
  const float* sa1_win     = (const float*)d_in[2];
  const float* sa1_bin     = (const float*)d_in[3];
  const float* sa1_wout    = (const float*)d_in[4];
  const float* sa1_bout    = (const float*)d_in[5];
  const float* sa2_win     = (const float*)d_in[6];
  const float* sa2_bin     = (const float*)d_in[7];
  const float* sa2_wout    = (const float*)d_in[8];
  const float* sa2_bout    = (const float*)d_in[9];
  const float* ca_win      = (const float*)d_in[10];
  const float* ca_bin      = (const float*)d_in[11];
  const float* ca_wout     = (const float*)d_in[12];
  const float* ca_bout     = (const float*)d_in[13];
  const float* ff_w1       = (const float*)d_in[14];
  const float* ff_b1       = (const float*)d_in[15];
  const float* ff_w2       = (const float*)d_in[16];
  const float* ff_b2       = (const float*)d_in[17];
  const float* query_embed = (const float*)d_in[18];
  const float* fc_rot_w    = (const float*)d_in[19];
  const float* fc_rot_b    = (const float*)d_in[20];
  const float* fc_trans_w  = (const float*)d_in[21];
  const float* fc_trans_b  = (const float*)d_in[22];

  const int Bv = in_sizes[0] / (S * E);

  // Carve bf16 weight images out of the workspace.
  __bf16* ws = (__bf16*)d_ws;
  size_t off = 0;
  auto carve = [&](size_t n) { __bf16* pp = ws + off; off += n; return pp; };
  const size_t N_WIN  = (size_t)2 * L * 3 * E * E;
  const size_t N_WOUT = (size_t)2 * L * E * E;
  const size_t N_W1   = (size_t)2 * L * FFD * E;
  const size_t N_W2   = (size_t)2 * L * E * FFD;

  __bf16* w_sa1_in  = carve(N_WIN);
  __bf16* w_sa1_out = carve(N_WOUT);
  __bf16* w_sa2_in  = carve(N_WIN);
  __bf16* w_sa2_out = carve(N_WOUT);
  __bf16* w_ca_in   = carve(N_WIN);
  __bf16* w_ca_out  = carve(N_WOUT);
  __bf16* w_ff1     = carve(N_W1);
  __bf16* w_ff2     = carve(N_W2);

  auto cvt = [&](const float* s, __bf16* d, size_t n) {
    const int th = 256;
    const int bl = (int)((n + th - 1) / th);
    f32_to_bf16_kernel<<<bl, th, 0, stream>>>(s, d, (int)n);
  };
  cvt(sa1_win,  w_sa1_in,  N_WIN);
  cvt(sa1_wout, w_sa1_out, N_WOUT);
  cvt(sa2_win,  w_sa2_in,  N_WIN);
  cvt(sa2_wout, w_sa2_out, N_WOUT);
  cvt(ca_win,   w_ca_in,   N_WIN);
  cvt(ca_wout,  w_ca_out,  N_WOUT);
  cvt(ff_w1,    w_ff1,     N_W1);
  cvt(ff_w2,    w_ff2,     N_W2);

  Params prm;
  prm.feature1 = feature1;   prm.feature2 = feature2;   prm.query_embed = query_embed;
  prm.w_sa1_in = w_sa1_in;   prm.w_sa1_out = w_sa1_out;
  prm.w_sa2_in = w_sa2_in;   prm.w_sa2_out = w_sa2_out;
  prm.w_ca_in  = w_ca_in;    prm.w_ca_out  = w_ca_out;
  prm.w_ff1    = w_ff1;      prm.w_ff2     = w_ff2;
  prm.b_sa1_in = sa1_bin;    prm.b_sa1_out = sa1_bout;
  prm.b_sa2_in = sa2_bin;    prm.b_sa2_out = sa2_bout;
  prm.b_ca_in  = ca_bin;     prm.b_ca_out  = ca_bout;
  prm.b_ff1    = ff_b1;      prm.b_ff2     = ff_b2;
  prm.fc_rot_w = fc_rot_w;   prm.fc_rot_b  = fc_rot_b;
  prm.fc_trans_w = fc_trans_w; prm.fc_trans_b = fc_trans_b;
  prm.out = (float*)d_out;

  const dim3 grid(Bv / ELEMS_PER_BLOCK, 2);
  const size_t shmem = sizeof(WaveBuf) * WAVES_PER_BLOCK;
  pose_transformer_kernel<<<grid, THREADS, shmem, stream>>>(prm);
}
